// AttentionBlock_32804960207124
// MI455X (gfx1250) — matrix operand order repaired and checked by reading, not compiled, not run on hardware
//
#include <hip/hip_runtime.h>

// ---------------- types ----------------
typedef __attribute__((ext_vector_type(16))) __bf16 v16bf;
typedef __attribute__((ext_vector_type(8)))  __bf16 v8bf;
typedef __attribute__((ext_vector_type(8)))  float  v8f;

#define DIM 512
#define L 2048
#define B 4
#define NHEADS 8
#define HDIM 64
#define NGROUPS 32
// 0.125 * log2(e): fold softmax scale + base-2 exp into q
#define QSCALE 0.18033688011112043f

// LDS row strides (bf16 units): 16B-aligned, conflict-free bank offsets
#define KSTR 88   // 176 B/row -> bank step 44
#define VSTR 56   // 112 B/row -> bank step 28

static __device__ __forceinline__ v8f wmma_bf16(v16bf a, v16bf b, v8f c) {
    return __builtin_amdgcn_wmma_f32_16x16x32_bf16(false, a, false, b, (short)0, c, false, false);
}

static __device__ __forceinline__ v16bf ld_frag(const __bf16* p0, const __bf16* p1) {
    v8bf lo = *(const v8bf*)p0;
    v8bf hi = *(const v8bf*)p1;
    return __builtin_shufflevector(lo, hi, 0,1,2,3,4,5,6,7,8,9,10,11,12,13,14,15);
}
// A-frag (16x32 K-chunk): elems i at K = half*16 + i
static __device__ __forceinline__ v16bf ld_a(const __bf16* row, int c, int half) {
    return ld_frag(row + c + half * 16, row + c + half * 16 + 8);
}
// B-frag (32x16 from [N,K] row-major): elems i at K = half*16 + i
static __device__ __forceinline__ v16bf ld_b(const __bf16* row, int c, int half) {
    return ld_frag(row + c + half * 16, row + c + half * 16 + 8);
}

static __device__ __forceinline__ unsigned pk_bf16(float lo, float hi) {
    __bf16 l = (__bf16)lo, h = (__bf16)hi;
    unsigned short lu = __builtin_bit_cast(unsigned short, l);
    unsigned short hu = __builtin_bit_cast(unsigned short, h);
    return ((unsigned)hu << 16) | (unsigned)lu;
}

// async copy 16B global -> LDS (gfx1250, tracked by ASYNCcnt)
static __device__ __forceinline__ void async_cp16(const __bf16* g, __bf16* l) {
    unsigned loff = (unsigned)(size_t)(void*)l;   // LDS aperture: low 32 bits = LDS offset
    asm volatile("global_load_async_to_lds_b128 %0, %1, off"
                 :: "v"(loff), "v"(g) : "memory");
}
static __device__ __forceinline__ void wait_async_2() {
    asm volatile("s_wait_asynccnt 2" ::: "memory");
}
static __device__ __forceinline__ void wait_async_0() {
    asm volatile("s_wait_asynccnt 0" ::: "memory");
}

// ---------------- K0: weight fp32 -> bf16 ----------------
__global__ void cvt_weights(const float* __restrict__ qw, const float* __restrict__ pw,
                            __bf16* __restrict__ qwb, __bf16* __restrict__ pwb) {
    int i = blockIdx.x * 256 + threadIdx.x;
    if (i < 3 * DIM * DIM) qwb[i] = (__bf16)qw[i];
    if (i < DIM * DIM)     pwb[i] = (__bf16)pw[i];
}

// ---------------- K1: GroupNorm stats ----------------
__global__ void gn_stats(const float* __restrict__ x, float* __restrict__ stats) {
    int bg = blockIdx.x;
    int b = bg >> 5, g = bg & 31;
    const float* base = x + ((size_t)b * DIM + g * 16) * L;
    float s = 0.f, ss = 0.f;
    for (int i = threadIdx.x; i < 16 * L; i += 256) {
        float v = base[i];
        s += v; ss += v * v;
    }
    #pragma unroll
    for (int o = 16; o; o >>= 1) {
        s  += __shfl_xor(s,  o, 32);
        ss += __shfl_xor(ss, o, 32);
    }
    __shared__ float rs[8], rss[8];
    int wid = threadIdx.x >> 5, lane = threadIdx.x & 31;
    if (lane == 0) { rs[wid] = s; rss[wid] = ss; }
    __syncthreads();
    if (threadIdx.x == 0) {
        float S = 0.f, SS = 0.f;
        #pragma unroll
        for (int i = 0; i < 8; i++) { S += rs[i]; SS += rss[i]; }
        float mean = S * (1.0f / (16.0f * L));
        float var  = SS * (1.0f / (16.0f * L)) - mean * mean;
        stats[bg * 2]     = mean;
        stats[bg * 2 + 1] = rsqrtf(var + 1e-5f);
    }
}

// ---------------- K2: normalize+affine -> hT [B, L, C] bf16 ----------------
__global__ void gn_apply(const float* __restrict__ x, const float* __restrict__ w,
                         const float* __restrict__ bb, const float* __restrict__ stats,
                         __bf16* __restrict__ hT) {
    size_t idx = (size_t)blockIdx.x * 256 + threadIdx.x;
    int l = (int)(idx & (L - 1));
    int c = (int)((idx >> 11) & (DIM - 1));
    int b = (int)(idx >> 20);
    int bg = b * NGROUPS + (c >> 4);
    float mean = stats[bg * 2], rstd = stats[bg * 2 + 1];
    float o = (x[idx] - mean) * rstd * w[c] + bb[c];
    hT[(((size_t)b * L + l) << 9) + c] = (__bf16)o;
}

// ---------------- K3: QKV GEMM, 64x32 tile per wave (4x2 wmma blocking) ----------------
__global__ void __launch_bounds__(256, 1)
qkv_gemm(const __bf16* __restrict__ wq, const __bf16* __restrict__ hT,
         const float* __restrict__ bias,
         __bf16* __restrict__ qT, __bf16* __restrict__ kT, __bf16* __restrict__ vm) {
    int wid = threadIdx.x >> 5, lane = threadIdx.x & 31;
    int tile = blockIdx.x * 8 + wid;          // 4 * 24 * 64 super-tiles
    int b   = tile / (24 * 64);
    int rem = tile % (24 * 64);
    int ot  = rem / 64, lt = rem % 64;        // 64-row, 32-col super-tile
    int n = lane & 15, half = lane >> 4;

    const __bf16* abase = wq + (size_t)(ot * 64 + n) * DIM;             // rows at +i*16*DIM
    const __bf16* bbase = hT + ((size_t)b * L + lt * 32 + n) * DIM;     // rows at +j*16*DIM

    v8f acc[4][2] = {};
    #pragma unroll 4
    for (int c = 0; c < DIM; c += 32) {
        v16bf b0 = ld_b(bbase, c, half);
        v16bf b1 = ld_b(bbase + 16 * DIM, c, half);
        #pragma unroll
        for (int i = 0; i < 4; i++) {
            v16bf a = ld_a(abase + i * 16 * DIM, c, half);
            acc[i][0] = wmma_bf16(a, b0, acc[i][0]);
            acc[i][1] = wmma_bf16(a, b1, acc[i][1]);
        }
    }

    // 64-row block = one head of one segment (uniform per wave)
    int seg = ot >> 3, hh = ot & 7;
    size_t bh = (size_t)(b * NHEADS + hh);
    #pragma unroll
    for (int i = 0; i < 4; i++) {
        #pragma unroll
        for (int j = 0; j < 2; j++) {
            #pragma unroll
            for (int r = 0; r < 8; r++) {
                int d = i * 16 + r + 8 * half;
                int l = lt * 32 + j * 16 + n;
                float v = acc[i][j][r] + bias[seg * DIM + hh * HDIM + d];
                if (seg == 0)      qT[(bh * L + l) * HDIM + d] = (__bf16)(v * QSCALE);
                else if (seg == 1) kT[(bh * L + l) * HDIM + d] = (__bf16)v;
                else               vm[(bh * HDIM + d) * L + l] = (__bf16)v;
            }
        }
    }
}

// ---------------- K4: flash attention; K/V staged in LDS via async copies ----------------
// block = 8 waves, all on the same (b,h); wave w handles query tile qt0+w.
__global__ void __launch_bounds__(256, 1)
attn_flash(const __bf16* __restrict__ qT, const __bf16* __restrict__ kT,
           const __bf16* __restrict__ vm, __bf16* __restrict__ aT) {
    __shared__ __align__(16) __bf16 kbuf[2][32 * KSTR];
    __shared__ __align__(16) __bf16 vbuf[2][64 * VSTR];

    int wid = threadIdx.x >> 5, lane = threadIdx.x & 31;
    int bhid = blockIdx.x >> 4;               // 0..31 = (b,h)
    int qt   = (blockIdx.x & 15) * 8 + wid;   // 0..127
    int b = bhid >> 3, h = bhid & 7;
    int n = lane & 15, half = lane >> 4;
    size_t bh = (size_t)(b * NHEADS + h);

    const __bf16* kbase = kT + bh * (size_t)L * HDIM;
    const __bf16* vbase = vm + bh * (size_t)HDIM * L;

    // loop-invariant q B-fragments
    const __bf16* qrow = qT + (bh * L + qt * 16 + n) * HDIM;
    v16bf bq0 = ld_b(qrow, 0, half);
    v16bf bq1 = ld_b(qrow, 32, half);

    int t = threadIdx.x;
    int kr = t >> 3, kc = t & 7;              // K tile: 32 rows x 128B
    int vr = t >> 2, vc = t & 3;              // V tile: 64 rows x 64B

    // prologue: stage kb=0 into buffer 0
    async_cp16(kbase + (size_t)kr * HDIM + kc * 8, &kbuf[0][kr * KSTR + kc * 8]);
    async_cp16(vbase + (size_t)vr * L + vc * 8,    &vbuf[0][vr * VSTR + vc * 8]);

    v8f o0 = {}, o1 = {}, o2 = {}, o3 = {};
    float mrun = -3.0e38f, srun = 0.f;

    #pragma unroll 2
    for (int it = 0; it < L / 32; it++) {
        int cur = it & 1;
        if (it < L / 32 - 1) {
            int kb2 = (it + 1) * 32;
            async_cp16(kbase + (size_t)(kb2 + kr) * HDIM + kc * 8, &kbuf[cur ^ 1][kr * KSTR + kc * 8]);
            async_cp16(vbase + (size_t)vr * L + kb2 + vc * 8,      &vbuf[cur ^ 1][vr * VSTR + vc * 8]);
            wait_async_2();
        } else {
            wait_async_0();
        }
        __syncthreads();

        const __bf16* kk = kbuf[cur];
        const __bf16* vv = vbuf[cur];

        // S^T = K^T * Q (two 16-key sub-tiles; K-dim d=64 -> 2 wmma each)
        const __bf16* krow0 = kk + n * KSTR;
        const __bf16* krow1 = kk + (16 + n) * KSTR;
        v8f z = {};
        v8f s0 = wmma_bf16(ld_a(krow0, 0, half), bq0, z);
        s0     = wmma_bf16(ld_a(krow0, 32, half), bq1, s0);
        v8f s1 = wmma_bf16(ld_a(krow1, 0, half), bq0, z);
        s1     = wmma_bf16(ld_a(krow1, 32, half), bq1, s1);

        // online softmax over keys (base-2 domain; q pre-scaled)
        float tmax = s0[0];
        #pragma unroll
        for (int r = 0; r < 8; r++) { tmax = fmaxf(tmax, s0[r]); tmax = fmaxf(tmax, s1[r]); }
        tmax = fmaxf(tmax, __shfl_xor(tmax, 16, 32));
        float mnew = fmaxf(mrun, tmax);
        float alpha = __builtin_amdgcn_exp2f(mrun - mnew);
        float p0[8], p1[8], rsum = 0.f;
        #pragma unroll
        for (int r = 0; r < 8; r++) {
            p0[r] = __builtin_amdgcn_exp2f(s0[r] - mnew);
            p1[r] = __builtin_amdgcn_exp2f(s1[r] - mnew);
            rsum += p0[r] + p1[r];
        }
        rsum += __shfl_xor(rsum, 16, 32);
        srun = srun * alpha + rsum;
        mrun = mnew;
        o0 *= alpha; o1 *= alpha; o2 *= alpha; o3 *= alpha;

        // build P^T B-fragment (keys = K dim, queries = N): cross-half exchange
        union { v16bf v; unsigned u[8]; } bp;
        #pragma unroll
        for (int j = 0; j < 4; j++) {
            unsigned k0 = pk_bf16(p0[2 * j], p0[2 * j + 1]);
            unsigned k1 = pk_bf16(p1[2 * j], p1[2 * j + 1]);
            unsigned recv = __shfl_xor(half ? k0 : k1, 16, 32);
            bp.u[j]     = half ? recv : k0;
            bp.u[4 + j] = half ? k1 : recv;
        }

        // O += V * P^T  (4 d-tiles of 16) from LDS
        o0 = wmma_bf16(ld_a(vv + (0 * 16 + n) * VSTR, 0, half), bp.v, o0);
        o1 = wmma_bf16(ld_a(vv + (1 * 16 + n) * VSTR, 0, half), bp.v, o1);
        o2 = wmma_bf16(ld_a(vv + (2 * 16 + n) * VSTR, 0, half), bp.v, o2);
        o3 = wmma_bf16(ld_a(vv + (3 * 16 + n) * VSTR, 0, half), bp.v, o3);

        __syncthreads();   // protect buffer reuse at it+1's prefetch
    }

    float inv = 1.0f / srun;
    __bf16* orow = aT + (((size_t)b * L + qt * 16 + n) << 9) + h * HDIM;
    v8f oo[4] = { o0, o1, o2, o3 };
    #pragma unroll
    for (int dt = 0; dt < 4; dt++) {
        #pragma unroll
        for (int j = 0; j < 4; j++) {
            unsigned pkv = pk_bf16(oo[dt][2 * j] * inv, oo[dt][2 * j + 1] * inv);
            *(unsigned*)(orow + dt * 16 + 8 * half + 2 * j) = pkv;
        }
    }
}

// ---------------- K5: proj GEMM (4x2 blocking) + bias + residual ----------------
__global__ void __launch_bounds__(256, 1)
proj_gemm(const __bf16* __restrict__ pw, const __bf16* __restrict__ aT,
          const float* __restrict__ pb, const float* __restrict__ x,
          float* __restrict__ out) {
    int wid = threadIdx.x >> 5, lane = threadIdx.x & 31;
    int tile = blockIdx.x * 8 + wid;          // 4 * 8 * 64 super-tiles
    int b   = tile / (8 * 64);
    int rem = tile % (8 * 64);
    int ot  = rem / 64, lt = rem % 64;
    int n = lane & 15, half = lane >> 4;

    const __bf16* abase = pw + (size_t)(ot * 64 + n) * DIM;
    const __bf16* bbase = aT + ((size_t)b * L + lt * 32 + n) * DIM;

    v8f acc[4][2] = {};
    #pragma unroll 4
    for (int c = 0; c < DIM; c += 32) {
        v16bf b0 = ld_b(bbase, c, half);
        v16bf b1 = ld_b(bbase + 16 * DIM, c, half);
        #pragma unroll
        for (int i = 0; i < 4; i++) {
            v16bf a = ld_a(abase + i * 16 * DIM, c, half);
            acc[i][0] = wmma_bf16(a, b0, acc[i][0]);
            acc[i][1] = wmma_bf16(a, b1, acc[i][1]);
        }
    }

    #pragma unroll
    for (int i = 0; i < 4; i++) {
        #pragma unroll
        for (int j = 0; j < 2; j++) {
            #pragma unroll
            for (int r = 0; r < 8; r++) {
                int o = ot * 64 + i * 16 + r + 8 * half;
                int l = lt * 32 + j * 16 + n;
                size_t idx = ((size_t)b * DIM + o) * L + l;
                out[idx] = acc[i][j][r] + pb[o] + x[idx];
            }
        }
    }
}

// ---------------- launch ----------------
extern "C" void kernel_launch(void* const* d_in, const int* in_sizes, int n_in,
                              void* d_out, int out_size, void* d_ws, size_t ws_size,
                              hipStream_t stream) {
    const float* x     = (const float*)d_in[0];
    const float* nw    = (const float*)d_in[1];
    const float* nb    = (const float*)d_in[2];
    const float* qkvw  = (const float*)d_in[3];
    const float* qkvb  = (const float*)d_in[4];
    const float* projw = (const float*)d_in[5];
    const float* projb = (const float*)d_in[6];
    float* out = (float*)d_out;

    char* ws = (char*)d_ws;
    size_t off = 0;
    auto take = [&](size_t bytes) {
        void* p = ws + off;
        off += (bytes + 255) & ~(size_t)255;
        return p;
    };
    __bf16* qwb   = (__bf16*)take((size_t)3 * DIM * DIM * 2);
    __bf16* pwb   = (__bf16*)take((size_t)DIM * DIM * 2);
    float*  stats = (float*) take((size_t)B * NGROUPS * 2 * 4);
    __bf16* hT    = (__bf16*)take((size_t)B * L * DIM * 2);
    __bf16* qT    = (__bf16*)take((size_t)B * L * DIM * 2);
    __bf16* kT    = (__bf16*)take((size_t)B * L * DIM * 2);
    __bf16* vm    = (__bf16*)take((size_t)B * L * DIM * 2);
    __bf16* aT    = (__bf16*)take((size_t)B * L * DIM * 2);

    cvt_weights<<<(3 * DIM * DIM) / 256, 256, 0, stream>>>(qkvw, projw, qwb, pwb);
    gn_stats   <<<B * NGROUPS, 256, 0, stream>>>(x, stats);
    gn_apply   <<<((size_t)B * DIM * L) / 256, 256, 0, stream>>>(x, nw, nb, stats, hT);
    qkv_gemm   <<<(B * 24 * 64) / 8, 256, 0, stream>>>(qwb, hT, qkvb, qT, kT, vm);
    attn_flash <<<(B * NHEADS * 16), 256, 0, stream>>>(qT, kT, vm, aT);
    proj_gemm  <<<(B * 8 * 64) / 8, 256, 0, stream>>>(pwb, aT, projb, x, out);
}
